// DGCNN_semseg_16509854285878
// MI455X (gfx1250) — matrix operand order repaired and checked by reading, not compiled, not run on hardware
//
#include <hip/hip_runtime.h>
#include <hip/hip_bf16.h>

// ---------------------------------------------------------------------------
// Types for WMMA (gfx1250, wave32): A/B = 16 x bf16, C/D = 8 x f32
// ---------------------------------------------------------------------------
typedef __bf16   v16bf __attribute__((ext_vector_type(16)));
typedef float    v8f   __attribute__((ext_vector_type(8)));
typedef unsigned v4u   __attribute__((ext_vector_type(4)));
typedef int      v8i   __attribute__((ext_vector_type(8)));
typedef int      v4i   __attribute__((ext_vector_type(4)));

__device__ __forceinline__ __bf16 f2bf(float f) {
    union { float f; unsigned u; } v; v.f = f;
    unsigned r = v.u + 0x7FFFu + ((v.u >> 16) & 1u);   // round-to-nearest-even
    unsigned short h = (unsigned short)(r >> 16);
    union { unsigned short s; __bf16 b; } o; o.s = h;
    return o.b;
}

// pair layout helper: element (c, col) of a (Kpad x ld) bf16 matrix lives at
// dword (c>>1)*ld + col, halfword (c&1)
__device__ __forceinline__ long long bfp(int c, int col, int ld) {
    return ((long long)(c >> 1) * ld + col) * 2 + (c & 1);
}

// ---------------------------------------------------------------------------
// One-time weight prep: pre-swizzle W (O x Kreal, f32) into bf16 A-fragments.
// Fragment (ot, ks): 32 lanes x 16 bf16 (32B/lane, contiguous per lane).
// A layout (16x32 bf16): lane L -> row ot*16 + L%16;  vgpr j -> K = ks*32 +
// (L/16)*16 + j*2 (+1).   grid = (nks, otiles), block = 32
// ---------------------------------------------------------------------------
__global__ void prep_w(const float* __restrict__ W, int O, int Kreal,
                       __bf16* __restrict__ frag)
{
    const int lane = threadIdx.x & 31;
    const int ks = blockIdx.x, ot = blockIdx.y, nks = gridDim.x;
    const int orow = ot * 16 + (lane & 15);
    const int half = lane >> 4;
    __bf16* dst = frag + (((long long)ot * nks + ks) * 32 + lane) * 16;
#pragma unroll
    for (int j = 0; j < 8; ++j) {
        int ca = ks * 32 + half * 16 + j * 2;
        float w0 = 0.f, w1 = 0.f;
        if (orow < O) {
            if (ca     < Kreal) w0 = W[(long long)orow * Kreal + ca];
            if (ca + 1 < Kreal) w1 = W[(long long)orow * Kreal + ca + 1];
        }
        dst[2 * j]     = f2bf(w0);
        dst[2 * j + 1] = f2bf(w1);
    }
}

// ---------------------------------------------------------------------------
// GEMM inner core.  USELDS: A fragments were DMA'd into LDS by the TDM in
// (ks-major, ot-minor) order -> the NUMOT fragments of one k-step sit at
// constant offsets +0,+1024,+2048,+3072 bytes.  Otherwise A is read from the
// global (ot-major) fragment array.
// ---------------------------------------------------------------------------
template<int NUMOT, bool USELDS>
__device__ __forceinline__ void gemm_core(const unsigned* lds_a,
                                          const v16bf* wfbase,   // global, + lane
                                          const unsigned* in_dw, unsigned boff,
                                          int ldp, int nks, int lane,
                                          v8f (&acc)[4])
{
    if (USELDS) {
        const v16bf* la = (const v16bf*)lds_a;
        int fidx = lane;                       // (ks*NUMOT + t)*32 + lane
        for (int ks = 0; ks < nks; ++ks) {
            union { unsigned u[8]; v16bf v; } bm;
#pragma unroll
            for (int j = 0; j < 8; ++j) bm.u[j] = in_dw[boff + (unsigned)j * (unsigned)ldp];
            boff += 16u * (unsigned)ldp;
#pragma unroll
            for (int t = 0; t < NUMOT; ++t) {
                v16bf aw = la[fidx + t * 32];
                acc[t] = __builtin_amdgcn_wmma_f32_16x16x32_bf16(
                             false, aw, false, bm.v, (short)0, acc[t], false, false);
            }
            fidx += NUMOT * 32;
        }
    } else {
        for (int ks = 0; ks < nks; ++ks) {
            union { unsigned u[8]; v16bf v; } bm;
#pragma unroll
            for (int j = 0; j < 8; ++j) bm.u[j] = in_dw[boff + (unsigned)j * (unsigned)ldp];
            boff += 16u * (unsigned)ldp;
#pragma unroll
            for (int t = 0; t < NUMOT; ++t) {
                v16bf aw = wfbase[((long long)t * nks + ks) * 32];
                acc[t] = __builtin_amdgcn_wmma_f32_16x16x32_bf16(
                             false, aw, false, bm.v, (short)0, acc[t], false, false);
            }
        }
    }
}

// ---------------------------------------------------------------------------
// GEMM:  Out[b] = act( W * In[b] )
//   Wfrag : pre-swizzled bf16 A-fragments (see prep_w)
//   In    : bf16 pair layout, nks*32 rows x Ncols cols, row-pair pitch ldp
//   Out   : f32 (obf=0) or bf16 pair layout (obf=1)
//   act   : 0 = none, 1 = leaky_relu(0.2)
// Block = 128 threads (4 waves, one 16-col tile each -> 64 cols).  Each wave
// accumulates up to 4 o-tiles (64 rows), sharing one B fragment per k-step.
// useLds = 1: wave 0 DMAs the block's A-fragment region into LDS via the
// Tensor Data Mover as a 3-D tile (x = 1KB fragment, y = o-tile @ stride
// nks*1KB, z = k-step @ stride 1KB), which lands in LDS transposed to
// (ks-major, ot-minor) order for immediate-offset ds_load_b128 feeds.
// grid = (Ncols/64, ceil(O/64), B).  Ncols % 64 == 0 guaranteed.
// ---------------------------------------------------------------------------
__global__ void __launch_bounds__(128)
wmma_gemm(const __bf16* __restrict__ Wfrag,
          const __bf16* __restrict__ In,
          void* __restrict__ Out,
          int O, int nks, int Ncols,
          int ldp, long long inBS,
          int ldOut, long long outBS,
          int act, int obf, int useLds)
{
    __shared__ unsigned lds_a[10240];   // up to 4 o-tiles x 10 k-steps x 1KB

    const int tid  = threadIdx.x;
    const int lane = tid & 31;
    const int wave = tid >> 5;
    const int nt   = blockIdx.x * 4 + wave;
    const int otg  = blockIdx.y, b = blockIdx.z;
    const int col  = nt * 16 + (lane & 15);
    const int half = lane >> 4;
    const int ot0  = otg * 4;
    const int otAll = (O + 15) >> 4;
    const int numOt = (otAll - ot0) < 4 ? (otAll - ot0) : 4;

    if (useLds) {
        if (tid < 32) {
            // --- TDM: 3-D tile, transposing fragments to (ks, ot) order ---
            unsigned long long ga =
                (unsigned long long)(const void*)(Wfrag + (long long)ot0 * nks * 512);
            unsigned lo = (unsigned)(unsigned long long)(void*)lds_a;
            v4u g0;
            g0[0] = 1u;                                   // count = 1
            g0[1] = lo;                                   // lds_addr
            g0[2] = (unsigned)ga;                         // global_addr[31:0]
            g0[3] = ((unsigned)(ga >> 32) & 0x01FFFFFFu)  // global_addr[56:32]
                    | 0x80000000u;                        // type = 2 (image)
            v8i g1;
            g1[0] = 0x00020000;                           // data_size = 4B
            g1[1] = (int)(256u << 16);                    // tensor_dim0 = 256
            g1[2] = (int)((unsigned)numOt << 16);         // tensor_dim1 = numOt
            g1[3] = (int)(256u << 16);                    // tile_dim0 = 256 dwords
            g1[4] = (int)((unsigned)numOt                 // tile_dim1 = numOt
                          | ((unsigned)nks << 16));       // tile_dim2 = nks
            g1[5] = (int)((unsigned)nks * 256u);          // dim0_stride = nks*256
            g1[6] = (int)(256u << 16);                    // dim1_stride = 256
            g1[7] = 0;
            v4i g2 = {nks, 0, 0, 0};                      // tensor_dim2 = nks
            v4i g3 = {0, 0, 0, 0};
#if defined(__clang_major__) && __clang_major__ >= 23
            v8i g4 = {0, 0, 0, 0, 0, 0, 0, 0};
            __builtin_amdgcn_tensor_load_to_lds(g0, g1, g2, g3, g4, 0);
#else
            __builtin_amdgcn_tensor_load_to_lds(g0, g1, g2, g3, 0);
#endif
            __builtin_amdgcn_s_wait_tensorcnt(0);
        }
        __syncthreads();
    }

    const unsigned* in_dw = (const unsigned*)(In + (long long)b * inBS);
    const unsigned boff = (unsigned)(half * 8) * (unsigned)ldp + (unsigned)col;
    const v16bf* wfbase = (const v16bf*)Wfrag + (long long)ot0 * nks * 32 + lane;

    v8f acc[4] = {};
    if (useLds) {
        if (numOt == 1) gemm_core<1, true>(lds_a, wfbase, in_dw, boff, ldp, nks, lane, acc);
        else            gemm_core<4, true>(lds_a, wfbase, in_dw, boff, ldp, nks, lane, acc);
    } else {
        if (numOt == 1) gemm_core<1, false>(lds_a, wfbase, in_dw, boff, ldp, nks, lane, acc);
        else            gemm_core<4, false>(lds_a, wfbase, in_dw, boff, ldp, nks, lane, acc);
    }

    if (obf) {
        __bf16* ob = (__bf16*)Out + (long long)b * outBS;
#pragma unroll
        for (int t = 0; t < 4; ++t) {
            if (t >= numOt) break;
#pragma unroll
            for (int j = 0; j < 8; ++j) {
                int o = (ot0 + t) * 16 + j + 8 * half;
                if (o < O) {
                    float v = acc[t][j];
                    if (act == 1) v = v > 0.f ? v : 0.2f * v;
                    ob[bfp(o, col, ldOut)] = f2bf(v);
                }
            }
        }
    } else {
        float* ob = (float*)Out + (long long)b * outBS;
#pragma unroll
        for (int t = 0; t < 4; ++t) {
            if (t >= numOt) break;
#pragma unroll
            for (int j = 0; j < 8; ++j) {
                int o = (ot0 + t) * 16 + j + 8 * half;
                if (o < O) {
                    float v = acc[t][j];
                    if (act == 1) v = v > 0.f ? v : 0.2f * v;
                    ob[(long long)o * ldOut + col] = v;
                }
            }
        }
    }
}

// ---------------------------------------------------------------------------
// KNN: one 256-thread workgroup per query point.  neg_dist = 2*q.r - |q|^2 -
// |r|^2 in LDS, then K iterative argmax passes (ties -> lowest index, matching
// jax.lax.top_k).  Nref <= 4096, C <= 64.  grid = (M, B)
// ---------------------------------------------------------------------------
#define KNN_BLK 256
__global__ void knn_kernel(const float* __restrict__ q, int q_ld, long long q_bs,
                           const float* __restrict__ r, int r_ld, long long r_bs,
                           int C, int M, int Nref, int K, int* __restrict__ idx)
{
    __shared__ float sdist[4096];
    __shared__ float rv[KNN_BLK];
    __shared__ int   ri[KNN_BLK];
    __shared__ float qs[64];

    const int m = blockIdx.x, b = blockIdx.y, tid = threadIdx.x;
    const float* qb = q + (long long)b * q_bs;
    const float* rb = r + (long long)b * r_bs;

    if (tid < C) qs[tid] = qb[(long long)tid * q_ld + m];
    __syncthreads();

    float qq = 0.f;
    for (int c = 0; c < C; ++c) qq += qs[c] * qs[c];

    for (int n = tid; n < Nref; n += KNN_BLK) {
        float inner = 0.f, rr = 0.f;
        for (int c = 0; c < C; ++c) {
            float rvv = rb[(long long)c * r_ld + n];
            inner += qs[c] * rvv;
            rr    += rvv * rvv;
        }
        sdist[n] = 2.0f * inner - qq - rr;
    }
    __syncthreads();

    for (int s = 0; s < K; ++s) {
        float best = -3.402823466e38f; int bi = 0x7fffffff;
        for (int n = tid; n < Nref; n += KNN_BLK) {
            float d = sdist[n];
            if (d > best || (d == best && n < bi)) { best = d; bi = n; }
        }
        rv[tid] = best; ri[tid] = bi;
        __syncthreads();
        for (int stp = KNN_BLK / 2; stp > 0; stp >>= 1) {
            if (tid < stp) {
                float ov = rv[tid + stp]; int oi = ri[tid + stp];
                if (ov > rv[tid] || (ov == rv[tid] && oi < ri[tid])) {
                    rv[tid] = ov; ri[tid] = oi;
                }
            }
            __syncthreads();
        }
        if (tid == 0) {
            idx[((long long)b * M + m) * K + s] = ri[0];
            sdist[ri[0]] = -3.402823466e38f;
        }
        __syncthreads();
    }
}

// ---------------------------------------------------------------------------
// Edge feature gather -> bf16 pair layout ebuf (Kpad x P*32), col=(p-p0)*32+j
//   j < k, c <  C : x[c, idx[p,j]] - x[c, p]
//   j < k, c < 2C : x[c-C, p]       otherwise 0
// ---------------------------------------------------------------------------
__global__ void edge_gather_bf(const float* __restrict__ x, int x_ld, long long x_bs,
                               int b, const int* __restrict__ idx, int N, int k,
                               int C, int Kpad, int p0, int P,
                               __bf16* __restrict__ ebuf)
{
    const long long total = (long long)Kpad * P * 32;
    const int colcnt = P * 32;
    const float* xb = x + (long long)b * x_bs;
    for (long long t = (long long)blockIdx.x * blockDim.x + threadIdx.x;
         t < total; t += (long long)gridDim.x * blockDim.x) {
        int c    = (int)(t / colcnt);
        int colr = (int)(t % colcnt);
        int p = p0 + (colr >> 5);
        int j = colr & 31;
        float v = 0.f;
        if (j < k && c < 2 * C) {
            if (c < C) {
                int nb = idx[((long long)b * N + p) * k + j];
                v = xb[(long long)c * x_ld + nb] - xb[(long long)c * x_ld + p];
            } else {
                v = xb[(long long)(c - C) * x_ld + p];
            }
        }
        ebuf[bfp(c, colr, colcnt)] = f2bf(v);
    }
}

// out[c, p0+p] = max_{j<k} hsrc[c, p*32 + j]   (hsrc is 64 x P*32, f32)
__global__ void edge_reduce_max(const float* __restrict__ hsrc, int P, int k, int p0,
                                float* __restrict__ out, int o_ld)
{
    int t = blockIdx.x * blockDim.x + threadIdx.x;
    if (t >= 64 * P) return;
    int c = t / P, p = t % P;
    const float* row = hsrc + (long long)c * (P * 32) + (long long)p * 32;
    float m = row[0];
    for (int j = 1; j < k; ++j) m = fmaxf(m, row[j]);
    out[(long long)c * o_ld + p0 + p] = m;
}

// out[b,c,p] = max_{j<K} feats[b, c, idx[b,p,j]]
__global__ void pool_gather_max(const float* __restrict__ feats, int f_ld, long long f_bs,
                                const int* __restrict__ idx, int M, int K,
                                float* __restrict__ out, int o_ld, long long o_bs, int total)
{
    int t = blockIdx.x * blockDim.x + threadIdx.x;
    if (t >= total) return;
    int b = t / (64 * M); int r = t % (64 * M); int c = r / M; int p = r % M;
    const float* fb = feats + (long long)b * f_bs + (long long)c * f_ld;
    const int*   ib = idx + ((long long)b * M + p) * K;
    float m = fb[ib[0]];
    for (int j = 1; j < K; ++j) m = fmaxf(m, fb[ib[j]]);
    out[(long long)b * o_bs + (long long)c * o_ld + p] = m;
}

// x = relu(x + y), elementwise (same compact layout)
__global__ void add_relu(float* __restrict__ x, const float* __restrict__ y, long long n)
{
    long long t = (long long)blockIdx.x * blockDim.x + threadIdx.x;
    if (t < n) { float v = x[t] + y[t]; x[t] = v > 0.f ? v : 0.f; }
}

// f32 compact (4, C, N) -> bf16 pair layout (4, C/2 dword-rows, N)
__global__ void to_bfpairs(const float* __restrict__ in, __bf16* __restrict__ out,
                           int C, int N, int total)
{
    int t = blockIdx.x * blockDim.x + threadIdx.x;
    if (t >= total) return;
    int b = t / (C * N); int r = t % (C * N); int c = r / N; int p = r % N;
    out[(long long)b * C * N + bfp(c, p, N)] = f2bf(in[t]);
}

// xt[o] = max_n tmp[o, n]
__global__ void colmax(const float* __restrict__ tmp, int O, int N, float* __restrict__ xt)
{
    int o = blockIdx.x * blockDim.x + threadIdx.x;
    if (o >= O) return;
    const float* row = tmp + (long long)o * N;
    float m = row[0];
    for (int n = 1; n < N; ++n) m = fmaxf(m, row[n]);
    xt[o] = m;
}

// g[b,o] = max over 4 stages of xt[(s*4+b)*1024 + o]
__global__ void gmax4(const float* __restrict__ xt, float* __restrict__ g)
{
    int t = blockIdx.x * blockDim.x + threadIdx.x;
    if (t >= 4 * 1024) return;
    int b = t >> 10, o = t & 1023;
    float m = xt[(0 * 4 + b) * 1024 + o];
    for (int s = 1; s < 4; ++s) m = fmaxf(m, xt[(s * 4 + b) * 1024 + o]);
    g[t] = m;
}

// cat[b,c,p] (bf16 pairs): c<1024 -> g[b,c] (broadcast), else x3[b, c-1024, p]
__global__ void concat_bcast_bf(const float* __restrict__ g,
                                const float* __restrict__ x3, int x_ld, long long x_bs,
                                int M, __bf16* __restrict__ out, long long obs, int total)
{
    int t = blockIdx.x * blockDim.x + threadIdx.x;
    if (t >= total) return;
    int b = t / (1088 * M); int r = t % (1088 * M); int c = r / M; int p = r % M;
    float v = (c < 1024) ? g[b * 1024 + c]
                         : x3[(long long)b * x_bs + (long long)(c - 1024) * x_ld + p];
    out[(long long)b * obs + bfp(c, p, M)] = f2bf(v);
}

// cat[b,c,p] (bf16 pairs): c<C1 -> dec[b, c, idxU[b,p]] (unpool), else feat
__global__ void unpool_concat_bf(const float* __restrict__ dec, int dld, long long dbs,
                                 const int* __restrict__ idxU,
                                 const float* __restrict__ feat, int fld, long long fbs,
                                 int C1, int C2, int M, __bf16* __restrict__ out,
                                 long long obs, int total)
{
    int t = blockIdx.x * blockDim.x + threadIdx.x;
    if (t >= total) return;
    int Ct = C1 + C2;
    int b = t / (Ct * M); int r = t % (Ct * M); int c = r / M; int p = r % M;
    float v;
    if (c < C1) {
        int s = idxU[(long long)b * M + p];
        v = dec[(long long)b * dbs + (long long)c * dld + s];
    } else {
        v = feat[(long long)b * fbs + (long long)(c - C1) * fld + p];
    }
    out[(long long)b * obs + bfp(c, p, M)] = f2bf(v);
}

// node copy: out[b,c,p] = x[b, c, p] for c<3, p<m  (x is (4,9,4096))
__global__ void copy_node(const float* __restrict__ x, int m, float* __restrict__ out)
{
    int t = blockIdx.x * blockDim.x + threadIdx.x;
    if (t >= 4 * 3 * m) return;
    int b = t / (3 * m); int r = t % (3 * m); int c = r / m; int p = r % m;
    out[t] = x[(long long)b * 9 * 4096 + (long long)c * 4096 + p];
}

// ---------------------------------------------------------------------------
// Host orchestration
// ---------------------------------------------------------------------------
static inline int cdiv(int a, int b) { return (a + b - 1) / b; }

// edge_conv: knn over `in` features, then per-edge MLP via WMMA, max over k.
static void edge_conv_run(hipStream_t stream,
                          const float* in, int C, int ld, long long bs, int N, int k,
                          const __bf16* wf1, const __bf16* wf2,
                          float* out /* (4,64,N) compact */,
                          int* idx, __bf16* ebuf, __bf16* h1bf, float* h2f)
{
    dim3 kg(N, 4);
    knn_kernel<<<kg, KNN_BLK, 0, stream>>>(in, ld, bs, in, ld, bs, C, N, N, k, idx);

    const int Kpad = (2 * C + 31) & ~31;
    const int nks1 = Kpad / 32;
    const int CH   = N < 1024 ? N : 1024;
    for (int b = 0; b < 4; ++b) {
        for (int p0 = 0; p0 < N; p0 += CH) {
            int P = (N - p0) < CH ? (N - p0) : CH;
            int ncols = P * 32;
            long long tot = (long long)Kpad * ncols;
            edge_gather_bf<<<(int)((tot + 255) / 256), 256, 0, stream>>>(
                in, ld, bs, b, idx, N, k, C, Kpad, p0, P, ebuf);
            dim3 g1(ncols / 64, 1, 1);   // O = 64 -> one group of 4 o-tiles
            if (wf2) {
                wmma_gemm<<<g1, 128, 0, stream>>>(wf1, ebuf, h1bf, 64, nks1, ncols,
                                                  ncols, 0, ncols, 0, 1, 1, 1);
                wmma_gemm<<<g1, 128, 0, stream>>>(wf2, h1bf, h2f, 64, 2, ncols,
                                                  ncols, 0, ncols, 0, 1, 0, 1);
            } else {
                wmma_gemm<<<g1, 128, 0, stream>>>(wf1, ebuf, h2f, 64, nks1, ncols,
                                                  ncols, 0, ncols, 0, 1, 0, 1);
            }
            edge_reduce_max<<<cdiv(64 * P, 256), 256, 0, stream>>>(
                h2f, P, k, p0, out + (long long)b * 64 * N, N);
        }
    }
}

extern "C" void kernel_launch(void* const* d_in, const int* in_sizes, int n_in,
                              void* d_out, int out_size, void* d_ws, size_t ws_size,
                              hipStream_t stream)
{
    const float* x = (const float*)d_in[0];

    // --- workspace layout ---
    size_t off = 0;
    auto alloc = [&](size_t bytes) -> void* {
        void* p = (char*)d_ws + off;
        off += (bytes + 255) & ~(size_t)255;
        return p;
    };
    int*    idx    = (int*)   alloc((size_t)4 * 4096 * 20 * sizeof(int));
    float*  hb     = (float*) alloc((size_t)4 * 64 * 4096 * sizeof(float));
    float*  x0b    = (float*) alloc((size_t)4 * 64 * 4096 * sizeof(float));
    float*  x1b    = (float*) alloc((size_t)4 * 64 * 1024 * sizeof(float));
    float*  x2b    = (float*) alloc((size_t)4 * 64 * 256  * sizeof(float));
    float*  x3b    = (float*) alloc((size_t)4 * 64 * 64   * sizeof(float));
    float*  p1f    = (float*) alloc((size_t)4 * 64 * 1024 * sizeof(float));
    float*  p2f    = (float*) alloc((size_t)4 * 64 * 256  * sizeof(float));
    float*  p3f    = (float*) alloc((size_t)4 * 64 * 64   * sizeof(float));
    float*  xtb    = (float*) alloc((size_t)4 * 4 * 1024  * sizeof(float));
    float*  gb     = (float*) alloc((size_t)4 * 1024      * sizeof(float));
    __bf16* wfrag  = (__bf16*)alloc((size_t)2 * 1024 * 1024 * sizeof(__bf16));
    __bf16* ebuf   = (__bf16*)alloc((size_t)128 * 32768  * sizeof(__bf16));
    __bf16* h1bf   = (__bf16*)alloc((size_t)64 * 32768   * sizeof(__bf16));
    float*  h2f    = (float*) alloc((size_t)64 * 32768   * sizeof(float));
    __bf16* xbf    = (__bf16*)alloc((size_t)4 * 64 * 4096 * sizeof(__bf16));
    float*  pn_tmp = (float*) alloc((size_t)1024 * 4096  * sizeof(float));
    __bf16* catbf  = (__bf16*)alloc((size_t)4 * 320 * 4096 * sizeof(__bf16));
    float*  decb   = (float*) alloc((size_t)4 * 256 * 1024 * sizeof(float));
    __bf16* decbf  = (__bf16*)alloc((size_t)4 * 128 * 4096 * sizeof(__bf16));
    (void)ws_size; (void)in_sizes; (void)n_in; (void)out_size;

    // --- one-time weight fragment prep (bf16, A-fragment swizzled) ---
    size_t woff = 0;
    auto prep = [&](int widx, int O, int K) -> __bf16* {
        int Opad = (O + 15) & ~15, Kpad = (K + 31) & ~31;
        __bf16* p = wfrag + woff;
        woff += (size_t)Opad * Kpad;
        prep_w<<<dim3(Kpad / 32, Opad / 16), 32, 0, stream>>>(
            (const float*)d_in[widx], O, K, p);
        return p;
    };
    __bf16* wf_ec1_0  = prep(1, 64, 18);
    __bf16* wf_ec1_1  = prep(2, 64, 64);
    __bf16* wf_ec2_0  = prep(3, 64, 128);
    __bf16* wf_ec2_1  = prep(4, 64, 64);
    __bf16* wf_ec4_0  = prep(5, 64, 128);
    __bf16* wf_ec4_1  = prep(6, 64, 64);
    __bf16* wf_ec5_0  = prep(7, 64, 128);
    __bf16* wf_ec5_1  = prep(8, 64, 64);
    __bf16* wf_ec7    = prep(9, 64, 128);
    __bf16* wf_ec8    = prep(10, 64, 128);
    __bf16* wf_ec10   = prep(11, 64, 128);
    __bf16* wf_ec11   = prep(12, 64, 128);
    __bf16* wf_pn3    = prep(13, 1024, 64);
    __bf16* wf_pn6    = prep(14, 1024, 64);
    __bf16* wf_pn9    = prep(15, 1024, 64);
    __bf16* wf_pn12   = prep(16, 1024, 64);
    __bf16* wf_pn13   = prep(17, 256, 1088);
    __bf16* wf_pn14   = prep(18, 256, 320);
    __bf16* wf_pn15   = prep(19, 256, 320);
    __bf16* wf_pn16   = prep(20, 128, 320);
    __bf16* wf_conv17 = prep(21, 13, 128);

    const long long xbs = 9LL * 4096;

    // per-point MLP + column max -> xt[stage]   (O = 1024, K = 64)
    auto pn_stage = [&](const __bf16* wf, const float* xs, int N, int stage) {
        to_bfpairs<<<cdiv(4 * 64 * N, 256), 256, 0, stream>>>(xs, xbf, 64, N, 4 * 64 * N);
        for (int b = 0; b < 4; ++b) {
            dim3 g(N / 64, 16, 1);
            wmma_gemm<<<g, 128, 0, stream>>>(wf, xbf + (long long)b * 64 * N, pn_tmp,
                                             1024, 2, N, N, 0, N, 0, 1, 0, 1);
            colmax<<<cdiv(1024, 256), 256, 0, stream>>>(pn_tmp, 1024, N,
                                                        xtb + (stage * 4 + b) * 1024);
        }
    };

    // --- encoder level 0 (N = 4096) ---
    edge_conv_run(stream, x, 9, 4096, xbs, 4096, 20, wf_ec1_0, wf_ec1_1, hb,
                  idx, ebuf, h1bf, h2f);
    edge_conv_run(stream, hb, 64, 4096, 64LL * 4096, 4096, 20, wf_ec2_0, wf_ec2_1, x0b,
                  idx, ebuf, h1bf, h2f);
    pn_stage(wf_pn3, x0b, 4096, 0);

    // --- pool 0 -> 1 (m = 1024), feats from x0 ---
    {
        dim3 kg(1024, 4);
        knn_kernel<<<kg, KNN_BLK, 0, stream>>>(x, 4096, xbs, x, 4096, xbs, 3, 1024, 4096, 20, idx);
        int tot = 4 * 64 * 1024;
        pool_gather_max<<<cdiv(tot, 256), 256, 0, stream>>>(x0b, 4096, 64LL * 4096, idx,
                                                            1024, 20, p1f, 1024, 64LL * 1024, tot);
    }

    // --- encoder level 1 (N = 1024) ---
    edge_conv_run(stream, p1f, 64, 1024, 64LL * 1024, 1024, 20, wf_ec4_0, wf_ec4_1, hb,
                  idx, ebuf, h1bf, h2f);
    edge_conv_run(stream, hb, 64, 1024, 64LL * 1024, 1024, 20, wf_ec5_0, wf_ec5_1, x1b,
                  idx, ebuf, h1bf, h2f);
    pn_stage(wf_pn6, x1b, 1024, 1);
    add_relu<<<cdiv(4 * 64 * 1024, 256), 256, 0, stream>>>(x1b, p1f, 4LL * 64 * 1024);

    // --- pool 1 -> 2 (m = 256), feats from x1 ---
    {
        dim3 kg(256, 4);
        knn_kernel<<<kg, KNN_BLK, 0, stream>>>(x, 4096, xbs, x, 4096, xbs, 3, 256, 1024, 20, idx);
        int tot = 4 * 64 * 256;
        pool_gather_max<<<cdiv(tot, 256), 256, 0, stream>>>(x1b, 1024, 64LL * 1024, idx,
                                                            256, 20, p2f, 256, 64LL * 256, tot);
    }

    // --- encoder level 2 (N = 256, single-layer edge convs) ---
    edge_conv_run(stream, p2f, 64, 256, 64LL * 256, 256, 20, wf_ec7, nullptr, hb,
                  idx, ebuf, h1bf, h2f);
    edge_conv_run(stream, hb, 64, 256, 64LL * 256, 256, 20, wf_ec8, nullptr, x2b,
                  idx, ebuf, h1bf, h2f);
    pn_stage(wf_pn9, x2b, 256, 2);
    add_relu<<<cdiv(4 * 64 * 256, 256), 256, 0, stream>>>(x2b, p2f, 4LL * 64 * 256);

    // --- pool 2 -> 3 (m = 64), feats from x2 ---
    {
        dim3 kg(64, 4);
        knn_kernel<<<kg, KNN_BLK, 0, stream>>>(x, 4096, xbs, x, 4096, xbs, 3, 64, 256, 20, idx);
        int tot = 4 * 64 * 64;
        pool_gather_max<<<cdiv(tot, 256), 256, 0, stream>>>(x2b, 256, 64LL * 256, idx,
                                                            64, 20, p3f, 64, 64LL * 64, tot);
    }

    // --- encoder level 3 (N = 64, k = 10) ---
    edge_conv_run(stream, p3f, 64, 64, 64LL * 64, 64, 10, wf_ec10, nullptr, hb,
                  idx, ebuf, h1bf, h2f);
    edge_conv_run(stream, hb, 64, 64, 64LL * 64, 64, 10, wf_ec11, nullptr, x3b,
                  idx, ebuf, h1bf, h2f);
    pn_stage(wf_pn12, x3b, 64, 3);
    add_relu<<<cdiv(4 * 64 * 64, 256), 256, 0, stream>>>(x3b, p3f, 4LL * 64 * 64);

    // --- global feature ---
    gmax4<<<cdiv(4096, 256), 256, 0, stream>>>(xtb, gb);

    // --- decoder level 3: lrelu(w_pn13 * [g ; x3])  (1088 -> 256, M = 64) ---
    {
        int tot = 4 * 1088 * 64;
        concat_bcast_bf<<<cdiv(tot, 256), 256, 0, stream>>>(gb, x3b, 64, 64LL * 64, 64,
                                                            catbf, 1088LL * 64, tot);
        dim3 g(1, 4, 4);
        wmma_gemm<<<g, 128, 0, stream>>>(wf_pn13, catbf, decb, 256, 34, 64,
                                         64, 1088LL * 64, 64, 256LL * 64, 1, 0,
                                         0);   // nks too big for LDS staging
    }

    // --- unpool 3 -> 2, concat x2, pn14 (320 -> 256, M = 256) ---
    {
        dim3 kg(256, 4);
        knn_kernel<<<kg, KNN_BLK, 0, stream>>>(x, 4096, xbs, x, 4096, xbs, 3, 256, 64, 1, idx);
        int tot = 4 * 320 * 256;
        unpool_concat_bf<<<cdiv(tot, 256), 256, 0, stream>>>(decb, 64, 256LL * 64, idx,
                                                             x2b, 256, 64LL * 256,
                                                             256, 64, 256, catbf, 320LL * 256, tot);
        dim3 g(4, 4, 4);
        wmma_gemm<<<g, 128, 0, stream>>>(wf_pn14, catbf, decb, 256, 10, 256,
                                         256, 320LL * 256, 256, 256LL * 256, 1, 0, 1);
    }

    // --- unpool 2 -> 1, concat x1, pn15 (320 -> 256, M = 1024) ---
    {
        dim3 kg(1024, 4);
        knn_kernel<<<kg, KNN_BLK, 0, stream>>>(x, 4096, xbs, x, 4096, xbs, 3, 1024, 256, 1, idx);
        int tot = 4 * 320 * 1024;
        unpool_concat_bf<<<cdiv(tot, 256), 256, 0, stream>>>(decb, 256, 256LL * 256, idx,
                                                             x1b, 1024, 64LL * 1024,
                                                             256, 64, 1024, catbf, 320LL * 1024, tot);
        dim3 g(16, 4, 4);
        wmma_gemm<<<g, 128, 0, stream>>>(wf_pn15, catbf, decb, 256, 10, 1024,
                                         1024, 320LL * 1024, 1024, 256LL * 1024, 1, 0, 1);
    }

    // --- unpool 1 -> 0, concat x0, pn16 (320 -> 128, M = 4096, bf16 out) ---
    {
        dim3 kg(4096, 4);
        knn_kernel<<<kg, KNN_BLK, 0, stream>>>(x, 4096, xbs, x, 4096, xbs, 3, 4096, 1024, 1, idx);
        int tot = 4 * 320 * 4096;
        unpool_concat_bf<<<cdiv(tot, 256), 256, 0, stream>>>(decb, 1024, 256LL * 1024, idx,
                                                             x0b, 4096, 64LL * 4096,
                                                             256, 64, 4096, catbf, 320LL * 4096, tot);
        dim3 g(64, 2, 4);
        wmma_gemm<<<g, 128, 0, stream>>>(wf_pn16, catbf, decbf, 128, 10, 4096,
                                         4096, 320LL * 4096, 4096, 128LL * 4096, 1, 1, 1);
    }

    // --- final conv17 (128 -> 13), straight into d_out (f32) ---
    float* out_f = (float*)d_out;
    {
        dim3 g(64, 1, 4);
        wmma_gemm<<<g, 128, 0, stream>>>(wf_conv17, decbf, out_f, 13, 4, 4096,
                                         4096, 128LL * 4096, 4096, 13LL * 4096, 0, 0, 1);
    }

    // --- node outputs: prefixes of x[:, :3, :] ---
    float* out_n1 = out_f + 4 * 13 * 4096;
    float* out_n2 = out_n1 + 4 * 3 * 1024;
    float* out_n3 = out_n2 + 4 * 3 * 256;
    copy_node<<<cdiv(4 * 3 * 1024, 256), 256, 0, stream>>>(x, 1024, out_n1);
    copy_node<<<cdiv(4 * 3 * 256,  256), 256, 0, stream>>>(x, 256,  out_n2);
    copy_node<<<cdiv(4 * 3 * 64,   256), 256, 0, stream>>>(x, 64,   out_n3);
}
